// LSTMClassifier_5317169512908
// MI455X (gfx1250) — matrix operand order repaired and checked by reading, not compiled, not run on hardware
//
#include <hip/hip_runtime.h>
#include <hip/hip_bf16.h>

// ---------------------------------------------------------------------------
// LSTMClassifier on gfx1250 (MI455X).
// Key insight: only timestep t=2 reaches the output, so the 64-step LSTM
// collapses to 3 steps. Heavy matmuls (lin@w_lk : 12288x150x512, h@w_lr :
// 4096x128x512 twice) run on v_wmma_f32_16x16x32_f16 (f16 in / f32 accum).
// ---------------------------------------------------------------------------

typedef __attribute__((ext_vector_type(16))) _Float16 v16h;
typedef __attribute__((ext_vector_type(8)))  _Float16 v8h;
typedef __attribute__((ext_vector_type(8)))  float    v8f;

#define NB   4096          // batch
#define NT   3             // timesteps actually needed (0,1,2)
#define NF   37            // features
#define LDA  160           // lin K padded (150 -> 160)
#define NH   128           // hidden
#define NZ   512           // 4*NH gates
#define KT_XZ 5            // 160/32 K-steps for lin GEMM
#define KT_R  4            // 128/32 K-steps for recurrence GEMM

__device__ __forceinline__ float lrelu(float x) { return x >= 0.0f ? x : 0.2f * x; }
__device__ __forceinline__ float sigm(float x)  { return 1.0f / (1.0f + __expf(-x)); }

// ---------------------------------------------------------------------------
// Kernel 1: preprocess.  One thread per (t, b); builds lin[150] in f16,
// K-padded to 160 with zeros.  lin row index = t*4096 + b.
// ---------------------------------------------------------------------------
__global__ void prep_kernel(const float* __restrict__ feature,
                            const float* __restrict__ wm1, const float* __restrict__ bm1,
                            const float* __restrict__ wm2, const float* __restrict__ bm2,
                            const float* __restrict__ wt1, const float* __restrict__ bt1,
                            const float* __restrict__ wt2, const float* __restrict__ bt2,
                            const float* __restrict__ ws1, const float* __restrict__ bs1,
                            const float* __restrict__ ws2, const float* __restrict__ bs2,
                            _Float16* __restrict__ lin) {
    int idx = blockIdx.x * blockDim.x + threadIdx.x;
    if (idx >= NT * NB) return;
    int t = idx / NB;
    int b = idx - t * NB;

    const float* f = feature + ((size_t)b * 64 + t) * NF;
    float fv[NF];
#pragma unroll
    for (int k = 0; k < NF; ++k) fv[k] = f[k];

    float body[3] = { fv[33], fv[34], fv[35] };
    float env[6];
#pragma unroll
    for (int k = 0; k < 6; ++k) env[k] = fv[27 + k];
    float Pa = log1pf(fv[31]);

    // M branch: 3 -> 8 -> 16
    float m1[8];
#pragma unroll
    for (int j = 0; j < 8; ++j) {
        float s = bm1[j];
#pragma unroll
        for (int k = 0; k < 3; ++k) s += body[k] * wm1[k * 8 + j];
        m1[j] = lrelu(s);
    }
    float Mv[16];
#pragma unroll
    for (int j = 0; j < 16; ++j) {
        float s = bm2[j];
#pragma unroll
        for (int k = 0; k < 8; ++k) s += m1[k] * wm2[k * 16 + j];
        Mv[j] = lrelu(s);
    }

    // Tsk branch: 37 -> 8 -> 16 (abs(lrelu(.)))
    float t1[8];
#pragma unroll
    for (int j = 0; j < 8; ++j) {
        float s = bt1[j];
        for (int k = 0; k < NF; ++k) s += fv[k] * wt1[k * 8 + j];
        t1[j] = fabsf(lrelu(s));
    }
    float Tsk[16], Psk[16];
#pragma unroll
    for (int j = 0; j < 16; ++j) {
        float s = bt2[j];
#pragma unroll
        for (int k = 0; k < 8; ++k) s += t1[k] * wt2[k * 16 + j];
        Tsk[j] = fabsf(lrelu(s));
        Psk[j] = log1pf(Tsk[j]);
    }

    // S branch: s_in(58) -> 32 -> 64
    float sin_[58];
#pragma unroll
    for (int j = 0; j < 16; ++j) { sin_[j] = Mv[j]; sin_[16 + j] = Tsk[j]; sin_[32 + j] = Psk[j]; }
    sin_[48] = Pa;
#pragma unroll
    for (int j = 0; j < 6; ++j) sin_[49 + j] = env[j];
#pragma unroll
    for (int j = 0; j < 3; ++j) sin_[55 + j] = body[j];

    float s1[32];
#pragma unroll
    for (int j = 0; j < 32; ++j) {
        float s = bs1[j];
        for (int k = 0; k < 58; ++k) s += sin_[k] * ws1[k * 32 + j];
        s1[j] = lrelu(s);
    }

    _Float16* L = lin + (size_t)idx * LDA;
    // lin = [M(16), Tsk(16), Psk(16), Pa(1), S(64), feature(37)], pad to 160
#pragma unroll
    for (int j = 0; j < 16; ++j) { L[j] = (_Float16)Mv[j]; L[16 + j] = (_Float16)Tsk[j]; L[32 + j] = (_Float16)Psk[j]; }
    L[48] = (_Float16)Pa;
    for (int j = 0; j < 64; ++j) {       // S: 32 -> 64
        float s = bs2[j];
#pragma unroll
        for (int k = 0; k < 32; ++k) s += s1[k] * ws2[k * 64 + j];
        L[49 + j] = (_Float16)lrelu(s);
    }
#pragma unroll
    for (int k = 0; k < NF; ++k) L[113 + k] = (_Float16)fv[k];
#pragma unroll
    for (int k = 150; k < LDA; ++k) L[k] = (_Float16)0.0f;
}

// ---------------------------------------------------------------------------
// Kernel 2: pack w_lk (150x512, zero-pad K to 160) and w_lr (128x512) into
// WMMA B-fragment layout: tile (kt,nt) holds 32x16 block; flat element
// lane*16+i = W[kt*32 + 8*(lane>=16) + (i&7) + 16*(i>=8)][nt*16 + (lane&15)].
// ---------------------------------------------------------------------------
__global__ void pack_kernel(const float* __restrict__ wlk, const float* __restrict__ wlr,
                            _Float16* __restrict__ wlkp, _Float16* __restrict__ wlrp) {
    int idx = blockIdx.x * blockDim.x + threadIdx.x;
    const int n_k = KT_XZ * 32 * 512;   // 81920
    const int n_r = KT_R * 32 * 512;    // 65536
    if (idx < n_k) {
        int tile = idx >> 9, e = idx & 511;
        int kt = tile >> 5, nt = tile & 31;
        int lane = e >> 4, i = e & 15;
        int krow = kt * 32 + ((lane >= 16) ? 8 : 0) + (i & 7) + ((i >> 3) << 4);
        int col  = nt * 16 + (lane & 15);
        wlkp[idx] = (krow < 150) ? (_Float16)wlk[krow * NZ + col] : (_Float16)0.0f;
    } else if (idx < n_k + n_r) {
        int j = idx - n_k;
        int tile = j >> 9, e = j & 511;
        int kt = tile >> 5, nt = tile & 31;
        int lane = e >> 4, i = e & 15;
        int krow = kt * 32 + ((lane >= 16) ? 8 : 0) + (i & 7) + ((i >> 3) << 4);
        int col  = nt * 16 + (lane & 15);
        wlrp[j] = (_Float16)wlr[krow * NZ + col];
    }
}

// ---------------------------------------------------------------------------
// WMMA fragment helpers (16-bit A layout per CDNA5 ISA 7.12.2):
// lane<16: M=lane,   K = base + {0..7, 16..23},  base = ks*32
// lane>=16: M=lane-16, K = base+8 + {0..7, 16..23}
// ---------------------------------------------------------------------------
union AFrag { v16h v; v8h h[2]; };

__device__ __forceinline__ v16h load_a(const _Float16* __restrict__ act, int row,
                                       int lda, int ks, int lhalf) {
    const _Float16* p = act + (size_t)row * lda + ks * 32 + (lhalf << 3);
    AFrag a;
    a.h[0] = *(const v8h*)(p);
    a.h[1] = *(const v8h*)(p + 16);
    return a.v;
}

// ---------------------------------------------------------------------------
// Kernel 3: xz = lin @ w_lk + b_l.  M=12288, K=160, N=512.
// Grid: 768 blocks (16 rows each) x 256 threads (8 waves, each wave 16x64).
// ---------------------------------------------------------------------------
__global__ void gemm_xz_kernel(const _Float16* __restrict__ lin,
                               const _Float16* __restrict__ wlkp,
                               const float* __restrict__ bl,
                               float* __restrict__ xz) {
    const int mt    = blockIdx.x;
    const int wave  = threadIdx.x >> 5;
    const int lane  = threadIdx.x & 31;
    const int row0  = mt * 16;
    const int nt0   = wave * 4;
    const int lhalf = lane >> 4;
    const int lm    = lane & 15;

    v8f acc[4] = {};
    for (int ks = 0; ks < KT_XZ; ++ks) {
        v16h a = load_a(lin, row0 + lm, LDA, ks, lhalf);
#pragma unroll
        for (int j = 0; j < 4; ++j) {
            const _Float16* bp = wlkp + ((size_t)(ks * 32 + nt0 + j) << 9) + (lane << 4);
            v16h bf = *(const v16h*)bp;
            acc[j] = __builtin_amdgcn_wmma_f32_16x16x32_f16(
                false, a, false, bf, (short)0, acc[j], false, false);
        }
    }
#pragma unroll
    for (int j = 0; j < 4; ++j) {
        int n = (nt0 + j) * 16 + lm;
        float bias = bl[n];
#pragma unroll
        for (int r = 0; r < 8; ++r) {
            int row = row0 + r + (lhalf << 3);
            xz[(size_t)row * NZ + n] = acc[j][r] + bias;
        }
    }
}

// ---------------------------------------------------------------------------
// Kernel 4: gates at t=0 (h=c=0  =>  z = xz[0], c = i*g, h = o*lrelu(c)).
// ---------------------------------------------------------------------------
__global__ void gates0_kernel(const float* __restrict__ xz,
                              float* __restrict__ cbuf,
                              _Float16* __restrict__ h16) {
    int idx = blockIdx.x * blockDim.x + threadIdx.x;
    if (idx >= NB * NH) return;
    int b = idx >> 7, n = idx & 127;
    const float* z = xz + (size_t)b * NZ;
    float iv = sigm(z[n]);
    float gv = lrelu(z[n + 256]);
    float ov = sigm(z[n + 384]);
    float cv = iv * gv;
    cbuf[idx] = cv;
    h16[idx]  = (_Float16)(ov * lrelu(cv));
}

// ---------------------------------------------------------------------------
// Kernel 5: one recurrence step (t=1 or t=2), fused:
//   z = xz[t] + h_{t-1} @ w_lr  (WMMA, 16x512 tile per block, LDS staging)
//   gates -> update c, h (in place; block owns its 16 rows exclusively).
// ---------------------------------------------------------------------------
__global__ void recur_kernel(_Float16* __restrict__ h16,       // in/out
                             const _Float16* __restrict__ wlrp,
                             const float* __restrict__ xz_t,   // xz + t*NB*NZ
                             float* __restrict__ cbuf,
                             float* __restrict__ h32,
                             int write_h32) {
    __shared__ float zs[16 * NZ];   // 32 KB
    const int mt    = blockIdx.x;
    const int wave  = threadIdx.x >> 5;
    const int lane  = threadIdx.x & 31;
    const int row0  = mt * 16;
    const int nt0   = wave * 4;
    const int lhalf = lane >> 4;
    const int lm    = lane & 15;

    v8f acc[4] = {};
    for (int ks = 0; ks < KT_R; ++ks) {
        v16h a = load_a(h16, row0 + lm, NH, ks, lhalf);
#pragma unroll
        for (int j = 0; j < 4; ++j) {
            const _Float16* bp = wlrp + ((size_t)(ks * 32 + nt0 + j) << 9) + (lane << 4);
            v16h bf = *(const v16h*)bp;
            acc[j] = __builtin_amdgcn_wmma_f32_16x16x32_f16(
                false, a, false, bf, (short)0, acc[j], false, false);
        }
    }
#pragma unroll
    for (int j = 0; j < 4; ++j) {
        int n = (nt0 + j) * 16 + lm;
#pragma unroll
        for (int r = 0; r < 8; ++r) {
            int m = r + (lhalf << 3);
            zs[m * NZ + n] = acc[j][r] + xz_t[(size_t)(row0 + m) * NZ + n];
        }
    }
    __syncthreads();

    for (int e = threadIdx.x; e < 16 * NH; e += 256) {
        int m = e >> 7, n = e & 127;
        int gb = row0 + m;
        const float* zr = zs + m * NZ;
        float iv = sigm(zr[n]);
        float fv = sigm(zr[n + 128]);
        float gv = lrelu(zr[n + 256]);
        float ov = sigm(zr[n + 384]);
        float cv = fv * cbuf[(size_t)gb * NH + n] + iv * gv;
        float hv = ov * lrelu(cv);
        cbuf[(size_t)gb * NH + n] = cv;
        h16[(size_t)gb * NH + n]  = (_Float16)hv;
        if (write_h32) h32[(size_t)gb * NH + n] = hv;
    }
}

// ---------------------------------------------------------------------------
// Kernel 6: head MLP 128->64->32->16->8->3, softmax, assemble outputs.
// d_out = [output (4096*3)] ++ [x (4096*13)].
// ---------------------------------------------------------------------------
__global__ void head_kernel(const float* __restrict__ h32,
                            const float* __restrict__ feature,
                            const float* __restrict__ wp1, const float* __restrict__ bp1,
                            const float* __restrict__ wp2, const float* __restrict__ bp2,
                            const float* __restrict__ wp3, const float* __restrict__ bp3,
                            const float* __restrict__ wp4, const float* __restrict__ bp4,
                            const float* __restrict__ wp5, const float* __restrict__ bp5,
                            float* __restrict__ out) {
    int b = blockIdx.x * blockDim.x + threadIdx.x;
    if (b >= NB) return;
    const float* h = h32 + (size_t)b * NH;

    float d1[64];
    for (int j = 0; j < 64; ++j) {
        float s = bp1[j];
        for (int k = 0; k < 128; ++k) s += h[k] * wp1[k * 64 + j];
        d1[j] = lrelu(s);
    }
    float d2[32];
    for (int j = 0; j < 32; ++j) {
        float s = bp2[j];
#pragma unroll
        for (int k = 0; k < 64; ++k) s += d1[k] * wp2[k * 32 + j];
        d2[j] = lrelu(s);
    }
    float d3[16];
    for (int j = 0; j < 16; ++j) {
        float s = bp3[j];
#pragma unroll
        for (int k = 0; k < 32; ++k) s += d2[k] * wp3[k * 16 + j];
        d3[j] = lrelu(s);
    }
    float d4[8];
#pragma unroll
    for (int j = 0; j < 8; ++j) {
        float s = bp4[j];
#pragma unroll
        for (int k = 0; k < 16; ++k) s += d3[k] * wp4[k * 8 + j];
        d4[j] = lrelu(s);
    }
    float d5[3];
#pragma unroll
    for (int j = 0; j < 3; ++j) {
        float s = bp5[j];
#pragma unroll
        for (int k = 0; k < 8; ++k) s += d4[k] * wp5[k * 3 + j];
        d5[j] = lrelu(s);
    }
    float mx = fmaxf(d5[0], fmaxf(d5[1], d5[2]));
    float e0 = __expf(d5[0] - mx), e1 = __expf(d5[1] - mx), e2 = __expf(d5[2] - mx);
    float inv = 1.0f / (e0 + e1 + e2);
    float p0 = e0 * inv, p1 = e1 * inv, p2 = e2 * inv;

    out[b * 3 + 0] = p0;
    out[b * 3 + 1] = p1;
    out[b * 3 + 2] = p2;

    float* x = out + NB * 3 + (size_t)b * 13;
    const float* f2 = feature + ((size_t)b * 64 + 2) * NF;
#pragma unroll
    for (int k = 0; k < 10; ++k) x[k] = f2[27 + k];
    x[10] = p0; x[11] = p1; x[12] = p2;
}

// ---------------------------------------------------------------------------
// Launch.  Workspace layout (bytes):
//   lin   f16 [12288][160]        @ 0          (3,932,160)
//   wlkp  f16 [5*32*512]          @ 3,932,160  (  163,840)
//   wlrp  f16 [4*32*512]          @ 4,096,000  (  131,072)
//   xz    f32 [3][4096][512]      @ 4,227,072  (25,165,824)
//   h16   f16 [4096][128]         @ 29,392,896 ( 1,048,576)
//   c     f32 [4096][128]         @ 30,441,472 ( 2,097,152)
//   h32   f32 [4096][128]         @ 32,538,624 ( 2,097,152)  -> total 34,635,776
// ---------------------------------------------------------------------------
extern "C" void kernel_launch(void* const* d_in, const int* in_sizes, int n_in,
                              void* d_out, int out_size, void* d_ws, size_t ws_size,
                              hipStream_t stream) {
    (void)in_sizes; (void)n_in; (void)out_size; (void)ws_size;
    const float* feature = (const float*)d_in[0];
    const float* wm1 = (const float*)d_in[1];  const float* bm1 = (const float*)d_in[2];
    const float* wm2 = (const float*)d_in[3];  const float* bm2 = (const float*)d_in[4];
    const float* wt1 = (const float*)d_in[5];  const float* bt1 = (const float*)d_in[6];
    const float* wt2 = (const float*)d_in[7];  const float* bt2 = (const float*)d_in[8];
    const float* ws1 = (const float*)d_in[9];  const float* bs1 = (const float*)d_in[10];
    const float* ws2 = (const float*)d_in[11]; const float* bs2 = (const float*)d_in[12];
    const float* wlk = (const float*)d_in[13]; const float* wlr = (const float*)d_in[14];
    const float* bl  = (const float*)d_in[15];
    const float* wp1 = (const float*)d_in[16]; const float* bp1 = (const float*)d_in[17];
    const float* wp2 = (const float*)d_in[18]; const float* bp2 = (const float*)d_in[19];
    const float* wp3 = (const float*)d_in[20]; const float* bp3 = (const float*)d_in[21];
    const float* wp4 = (const float*)d_in[22]; const float* bp4 = (const float*)d_in[23];
    const float* wp5 = (const float*)d_in[24]; const float* bp5 = (const float*)d_in[25];

    char* ws = (char*)d_ws;
    _Float16* lin  = (_Float16*)(ws);
    _Float16* wlkp = (_Float16*)(ws + 3932160);
    _Float16* wlrp = (_Float16*)(ws + 4096000);
    float*    xz   = (float*)   (ws + 4227072);
    _Float16* h16  = (_Float16*)(ws + 29392896);
    float*    cbuf = (float*)   (ws + 30441472);
    float*    h32  = (float*)   (ws + 32538624);
    float*    out  = (float*)d_out;

    prep_kernel<<<(NT * NB + 255) / 256, 256, 0, stream>>>(
        feature, wm1, bm1, wm2, bm2, wt1, bt1, wt2, bt2, ws1, bs1, ws2, bs2, lin);

    pack_kernel<<<(81920 + 65536 + 255) / 256, 256, 0, stream>>>(wlk, wlr, wlkp, wlrp);

    gemm_xz_kernel<<<(NT * NB) / 16, 256, 0, stream>>>(lin, wlkp, bl, xz);

    gates0_kernel<<<(NB * NH) / 256, 256, 0, stream>>>(xz, cbuf, h16);

    recur_kernel<<<NB / 16, 256, 0, stream>>>(h16, wlrp, xz + (size_t)1 * NB * NZ, cbuf, h32, 0);
    recur_kernel<<<NB / 16, 256, 0, stream>>>(h16, wlrp, xz + (size_t)2 * NB * NZ, cbuf, h32, 1);

    head_kernel<<<NB / 256, 256, 0, stream>>>(h32, feature,
        wp1, bp1, wp2, bp2, wp3, bp3, wp4, bp4, wp5, bp5, out);
}
